// ClassifierChainModel_45569603010941
// MI455X (gfx1250) — compile-verified, matrix operand order checked by reading, not run on hardware
//
#include <hip/hip_runtime.h>
#include <hip/hip_bf16.h>
#include <stdint.h>

// ---------------------------------------------------------------------------
// CDNA5 (gfx1250): bf16 WMMA GEMMs + register-resident WMMA LSTM
//  - async global->LDS staging (GLOBAL_LOAD_ASYNC_TO_LDS_B128) when available
//  - hardware v_tanh_f32 for all activations when available
// ---------------------------------------------------------------------------

typedef __attribute__((ext_vector_type(16))) __bf16 v16bf;
typedef __attribute__((ext_vector_type(8)))  float  v8f;
typedef __attribute__((ext_vector_type(4)))  unsigned int v4u;
typedef int v4i_vec __attribute__((vector_size(16)));   // matches builtin param

#define AS1 __attribute__((address_space(1)))
#define AS3 __attribute__((address_space(3)))

#if defined(__has_builtin)
#if __has_builtin(__builtin_amdgcn_global_load_async_to_lds_b128)
#define ASYNC_LDS 1
#endif
#if __has_builtin(__builtin_amdgcn_s_wait_asynccnt)
#define WAIT_ASYNC_BUILTIN 1
#endif
#if __has_builtin(__builtin_amdgcn_tanhf)
#define HW_TANH(x) __builtin_amdgcn_tanhf(x)
#elif __has_builtin(__builtin_amdgcn_tanh_f32)
#define HW_TANH(x) __builtin_amdgcn_tanh_f32(x)
#endif
#endif

union FragBF {
    v16bf v;
    v4u   u[2];
};

__device__ __forceinline__ uint16_t f2bf(float f) {
    uint32_t u = __float_as_uint(f);
    uint32_t r = (u + 0x7FFFu + ((u >> 16) & 1u)) >> 16;
    return (uint16_t)r;
}

__device__ __forceinline__ float fast_tanh(float x) {
#ifdef HW_TANH
    return HW_TANH(x);
#else
    x = fminf(fmaxf(x, -10.0f), 10.0f);
    const float e = __expf(2.0f * x);
    return (e - 1.0f) / (e + 1.0f);
#endif
}

__device__ __forceinline__ float sigf(float x) {
    return 0.5f + 0.5f * fast_tanh(0.5f * x);
}

#ifdef ASYNC_LDS
__device__ __forceinline__ void async_copy16(const uint16_t* gsrc,
                                             uint16_t* ldst) {
    __builtin_amdgcn_global_load_async_to_lds_b128(
        (AS1 v4i_vec*)gsrc, (AS3 v4i_vec*)ldst, 0, 0);
}
__device__ __forceinline__ void wait_async0() {
#ifdef WAIT_ASYNC_BUILTIN
    __builtin_amdgcn_s_wait_asynccnt(0);
#else
    asm volatile("s_wait_asynccnt 0" ::: "memory");
#endif
}
#endif

// ---------------------------------------------------------------------------
// Prep kernels
// ---------------------------------------------------------------------------

__global__ void embed_kernel(const int* __restrict__ ids,
                             const float* __restrict__ emb,
                             uint16_t* __restrict__ xpad) {
    int idx = blockIdx.x * blockDim.x + threadIdx.x;
    const int total = 64 * 1026 * 64;
    if (idx >= total) return;
    int e    = idx & 63;
    int rem  = idx >> 6;
    int srow = rem % 1026;
    int b    = rem / 1026;
    float val = 0.0f;
    if (srow >= 1 && srow <= 1024) {
        int id = ids[b * 1024 + (srow - 1)];
        val = emb[id * 64 + e];
    }
    xpad[idx] = f2bf(val);
}

__global__ void zero_a1_halo_kernel(uint16_t* __restrict__ a1pad) {
    int idx = blockIdx.x * blockDim.x + threadIdx.x;  // 64 * 2 * 128
    if (idx >= 64 * 2 * 128) return;
    int c = idx & 127;
    int w = (idx >> 7) & 1;
    int b = idx >> 8;
    a1pad[(size_t)b * 1026 * 128 + (w ? 1025 : 0) * 128 + c] = 0;
}

__global__ void convert_bf16_kernel(const float* __restrict__ in,
                                    uint16_t* __restrict__ out, int n) {
    int i = blockIdx.x * blockDim.x + threadIdx.x;
    if (i < n) out[i] = f2bf(in[i]);
}

__global__ void repack_conv1_kernel(const float* __restrict__ w,
                                    uint16_t* __restrict__ out) {
    int i = blockIdx.x * blockDim.x + threadIdx.x;  // 128*192
    if (i >= 128 * 192) return;
    int k = i % 192, m = i / 192;
    int kk = k / 64, e = k % 64;
    out[i] = f2bf(w[(m * 64 + e) * 3 + kk]);
}

__global__ void repack_conv2_kernel(const float* __restrict__ w,
                                    uint16_t* __restrict__ out) {
    int i = blockIdx.x * blockDim.x + threadIdx.x;  // 256*384
    if (i >= 256 * 384) return;
    int k = i % 384, m = i / 384;
    int kk = k / 128, c = k % 128;
    out[i] = f2bf(w[(m * 128 + c) * 3 + kk]);
}

__global__ void bias_sum_kernel(const float* __restrict__ bif,
                                const float* __restrict__ bhf,
                                const float* __restrict__ bib,
                                const float* __restrict__ bhb,
                                float* __restrict__ bf,
                                float* __restrict__ bb) {
    int i = blockIdx.x * blockDim.x + threadIdx.x;
    if (i < 256)       bf[i] = bif[i] + bhf[i];
    else if (i < 512)  bb[i - 256] = bib[i - 256] + bhb[i - 256];
}

// ---------------------------------------------------------------------------
// Generic WMMA bf16 GEMM with implicit im2col addressing.
//   C[m][n] = sum_k A[m][k] * Brow(n)[k] + bias[m]    (n = b*1024 + s)
//   Brow(n) = Bact + b*batchStride + s*rowStride      (length >= K, padded)
//   C index = b*cBatch + s*cRow + m
// A-tile (16 x K, contiguous) is staged into LDS once per block (async when
// available) and shared by the 4 waves; wave -> one 16x16 tile of C.
// ---------------------------------------------------------------------------
template <bool BF16OUT, bool RELU>
__global__ __launch_bounds__(128)
void gemm_wmma_kernel(const uint16_t* __restrict__ A,
                      const uint16_t* __restrict__ Bact,
                      const float* __restrict__ bias,
                      uint16_t* __restrict__ CoutBf,
                      float* __restrict__ CoutF,
                      int K, int rowStride, long long batchStride,
                      long long cBatch, long long cRow) {
    __shared__ __align__(16) uint16_t As[16 * 384];  // max K = 384

    const int lane = threadIdx.x & 31;
    const int wave = threadIdx.x >> 5;
    const int half = lane >> 4;    // 0: K{0..7,16..23}, 1: K{8..15,24..31}
    const int l16  = lane & 15;

    const int m0 = blockIdx.y * 16;
    const int n0 = (blockIdx.x * 4 + wave) * 16;

    // stage the block's A tile (rows m0..m0+15 = one contiguous 16*K chunk)
    {
        const uint16_t* ablk = A + (size_t)m0 * K;
        const int nchunks = (16 * K) >> 3;  // 8 bf16 per 16B chunk
#ifdef ASYNC_LDS
        for (int i = threadIdx.x; i < nchunks; i += 128)
            async_copy16(ablk + i * 8, As + i * 8);
        wait_async0();
#else
        for (int i = threadIdx.x; i < nchunks; i += 128)
            *(v4u*)(As + i * 8) = *(const v4u*)(ablk + i * 8);
#endif
        __syncthreads();
    }

    const int n = n0 + l16;
    const int b = n >> 10;          // / 1024
    const int s = n & 1023;

    const uint16_t* arow = As + (size_t)l16 * K;
    const uint16_t* brow = Bact + (size_t)b * batchStride + (size_t)s * rowStride;

    v8f acc = {0.f, 0.f, 0.f, 0.f, 0.f, 0.f, 0.f, 0.f};
    for (int k0 = 0; k0 < K; k0 += 32) {
        const int kb = k0 + half * 8;
        FragBF af, bfr;
        af.u[0]  = *(const v4u*)(arow + kb);
        af.u[1]  = *(const v4u*)(arow + kb + 16);
        bfr.u[0] = *(const v4u*)(brow + kb);
        bfr.u[1] = *(const v4u*)(brow + kb + 16);
        acc = __builtin_amdgcn_wmma_f32_16x16x32_bf16(
            false, af.v, false, bfr.v, (short)0, acc, false, false);
    }

#pragma unroll
    for (int r = 0; r < 8; ++r) {
        const int m = m0 + r + half * 8;
        float val = acc[r] + bias[m];
        if (RELU) val = fmaxf(val, 0.0f);
        const size_t ci = (size_t)b * cBatch + (size_t)s * cRow + m;
        if (BF16OUT) CoutBf[ci] = f2bf(val);
        else         CoutF[ci]  = val;
    }
}

// ---------------------------------------------------------------------------
// LSTM scan: one workgroup per direction, 8 waves.
// Gates G[b][g] = xg[s][b][g] + h[b][:] @ w_hh[g][:]^T, M=64(B) N=256(G) K=64(H)
// Each wave owns 2 (b-tile, h-tile) pairs; the four gate tiles (i,f,g,o) share
// an identical D-matrix lane/slot layout so the cell update is pure per-lane
// VALU (hardware tanh). c and running-max(h) live in registers; h is shared
// through LDS (bf16); w_hh is staged into LDS once (async when available).
// ---------------------------------------------------------------------------
__global__ __launch_bounds__(256)
void lstm_kernel(const float* __restrict__ xg,       // [2][S][B][256]
                 const uint16_t* __restrict__ whhF,  // [256][64] bf16
                 const uint16_t* __restrict__ whhB,
                 float* __restrict__ gmp) {          // [B][128]
    const int dir = blockIdx.x;
    const uint16_t* whh = dir ? whhB : whhF;

    __shared__ __align__(16) uint16_t hbuf[64 * 64];
    __shared__ __align__(16) uint16_t whs[256 * 64];

    const int tid = threadIdx.x;
    for (int i = tid; i < 64 * 64; i += 256) hbuf[i] = 0;
#ifdef ASYNC_LDS
    for (int i = tid; i < 2048; i += 256)   // 2048 x 16B = 32KB
        async_copy16(whh + i * 8, whs + i * 8);
    wait_async0();
#else
    for (int i = tid; i < 256 * 64; i += 256) whs[i] = whh[i];
#endif
    __syncthreads();

    const int lane = tid & 31, wave = tid >> 5;
    const int half = lane >> 4, l16 = lane & 15;

    float c[2][8], hreg[2][8], hmax[2][8];
    int m0p[2], h0p[2];
#pragma unroll
    for (int p = 0; p < 2; ++p) {
        const int t = wave * 2 + p;
        m0p[p] = (t & 3) * 16;
        h0p[p] = (t >> 2) * 16;
#pragma unroll
        for (int r = 0; r < 8; ++r) { c[p][r] = 0.f; hmax[p][r] = -3.0e38f; }
    }

    for (int step = 0; step < 1024; ++step) {
        const int s = dir ? (1023 - step) : step;
        const float* xgs = xg + ((size_t)dir * 1024 + s) * (64 * 256);

        for (int p = 0; p < 2; ++p) {
            const int m0 = m0p[p], h0 = h0p[p];
            v8f acc[4];
#pragma unroll
            for (int gi = 0; gi < 4; ++gi)
#pragma unroll
                for (int r = 0; r < 8; ++r)
                    acc[gi][r] = xgs[(size_t)(m0 + r + half * 8) * 256 +
                                     gi * 64 + h0 + l16];

#pragma unroll
            for (int k0 = 0; k0 < 64; k0 += 32) {
                const int kb = k0 + half * 8;
                FragBF af;
                af.u[0] = *(const v4u*)&hbuf[(m0 + l16) * 64 + kb];
                af.u[1] = *(const v4u*)&hbuf[(m0 + l16) * 64 + kb + 16];
#pragma unroll
                for (int gi = 0; gi < 4; ++gi) {
                    const int g = gi * 64 + h0 + l16;
                    FragBF bfr;
                    bfr.u[0] = *(const v4u*)&whs[g * 64 + kb];
                    bfr.u[1] = *(const v4u*)&whs[g * 64 + kb + 16];
                    acc[gi] = __builtin_amdgcn_wmma_f32_16x16x32_bf16(
                        false, af.v, false, bfr.v, (short)0, acc[gi],
                        false, false);
                }
            }

#pragma unroll
            for (int r = 0; r < 8; ++r) {
                const float ig = sigf(acc[0][r]);
                const float fg = sigf(acc[1][r]);
                const float gg = fast_tanh(acc[2][r]);
                const float og = sigf(acc[3][r]);
                const float cn = fg * c[p][r] + ig * gg;
                c[p][r] = cn;
                const float h = og * fast_tanh(cn);
                hreg[p][r] = h;
                hmax[p][r] = fmaxf(hmax[p][r], h);
            }
        }
        __syncthreads();  // all waves finished reading hbuf for this step
#pragma unroll
        for (int p = 0; p < 2; ++p)
#pragma unroll
            for (int r = 0; r < 8; ++r)
                hbuf[(m0p[p] + r + half * 8) * 64 + h0p[p] + l16] =
                    f2bf(hreg[p][r]);
        __syncthreads();  // h published for next step
    }

#pragma unroll
    for (int p = 0; p < 2; ++p)
#pragma unroll
        for (int r = 0; r < 8; ++r) {
            const int b = m0p[p] + r + half * 8;
            const int h = h0p[p] + l16;
            gmp[b * 128 + dir * 64 + h] = hmax[p][r];
        }
}

// ---------------------------------------------------------------------------
// Head: feature branch + weight-normed classifier chain. One thread per batch.
// ---------------------------------------------------------------------------
__device__ __forceinline__ float wn_logit(const float* __restrict__ v, float g,
                                          float b, const float* __restrict__ x,
                                          int din) {
    float nrm = 0.f, dot = 0.f;
    for (int k = 0; k < din; ++k) {
        nrm += v[k] * v[k];
        dot += v[k] * x[k];
    }
    return dot * (g * __frsqrt_rn(nrm)) + b;
}

__global__ void head_kernel(const float* __restrict__ gmp,
                            const float* __restrict__ feat,
                            const float* __restrict__ feat_w,
                            const float* __restrict__ feat_b,
                            const float* __restrict__ bn_g,
                            const float* __restrict__ bn_b,
                            const float* __restrict__ bn_m,
                            const float* __restrict__ bn_v,
                            const float* __restrict__ tox_v, const float* tox_g, const float* tox_b,
                            const float* __restrict__ ins_v, const float* ins_g, const float* ins_b,
                            const float* __restrict__ prof_v, const float* prof_g, const float* prof_b,
                            const float* __restrict__ thr_v, const float* thr_g, const float* thr_b,
                            const float* __restrict__ idh_v, const float* idh_g, const float* idh_b,
                            const float* __restrict__ sev_v, const float* sev_g, const float* sev_b,
                            float* __restrict__ out) {
    const int b = threadIdx.x;
    if (b >= 64) return;

    float base[165];
    for (int i = 0; i < 128; ++i) base[i] = gmp[b * 128 + i];
    for (int j = 0; j < 32; ++j) {
        float acc = feat_b[j];
        for (int k = 0; k < 16; ++k) acc += feat[b * 16 + k] * feat_w[j * 16 + k];
        acc = (acc - bn_m[j]) * __frsqrt_rn(bn_v[j] + 1e-5f) * bn_g[j] + bn_b[j];
        base[128 + j] = fmaxf(acc, 0.0f);
    }

    const float tox_l = wn_logit(tox_v, tox_g[0], tox_b[0], base, 160);
    const float tox_p = sigf(tox_l);
    base[160] = tox_p;

    float cat_l[4], cat_p[4];
    cat_l[0] = wn_logit(ins_v,  ins_g[0],  ins_b[0],  base, 161);
    cat_l[1] = wn_logit(prof_v, prof_g[0], prof_b[0], base, 161);
    cat_l[2] = wn_logit(thr_v,  thr_g[0],  thr_b[0],  base, 161);
    cat_l[3] = wn_logit(idh_v,  idh_g[0],  idh_b[0],  base, 161);
    for (int j = 0; j < 4; ++j) {
        cat_p[j] = sigf(cat_l[j]);
        base[161 + j] = cat_p[j];
    }

    const float sev_l = wn_logit(sev_v, sev_g[0], sev_b[0], base, 165);
    const float sev_p = sigf(sev_l);

    float* o = out + b * 12;
    o[0] = tox_l;
    o[1] = cat_l[0]; o[2] = cat_l[1]; o[3] = cat_l[2]; o[4] = cat_l[3];
    o[5] = sev_l;
    o[6] = tox_p;
    o[7] = cat_p[0]; o[8] = cat_p[1]; o[9] = cat_p[2]; o[10] = cat_p[3];
    o[11] = sev_p;
}

// ---------------------------------------------------------------------------
// Launch
// ---------------------------------------------------------------------------
extern "C" void kernel_launch(void* const* d_in, const int* in_sizes, int n_in,
                              void* d_out, int out_size, void* d_ws, size_t ws_size,
                              hipStream_t stream) {
    (void)in_sizes; (void)n_in; (void)out_size; (void)ws_size;

    const int* char_ids   = (const int*)d_in[0];
    const float* tox_feat = (const float*)d_in[1];
    const float* emb      = (const float*)d_in[2];
    const float* conv1_w  = (const float*)d_in[3];
    const float* conv1_b  = (const float*)d_in[4];
    const float* conv2_w  = (const float*)d_in[5];
    const float* conv2_b  = (const float*)d_in[6];
    const float* fc_w     = (const float*)d_in[7];
    const float* fc_b     = (const float*)d_in[8];
    const float* w_ih_f   = (const float*)d_in[9];
    const float* w_hh_f   = (const float*)d_in[10];
    const float* b_ih_f   = (const float*)d_in[11];
    const float* b_hh_f   = (const float*)d_in[12];
    const float* w_ih_b   = (const float*)d_in[13];
    const float* w_hh_b   = (const float*)d_in[14];
    const float* b_ih_b   = (const float*)d_in[15];
    const float* b_hh_b   = (const float*)d_in[16];
    const float* feat_w   = (const float*)d_in[17];
    const float* feat_b   = (const float*)d_in[18];
    const float* bn_g     = (const float*)d_in[19];
    const float* bn_b     = (const float*)d_in[20];
    const float* bn_m     = (const float*)d_in[21];
    const float* bn_v     = (const float*)d_in[22];
    const float* tox_v = (const float*)d_in[23]; const float* tox_g = (const float*)d_in[24]; const float* tox_bb = (const float*)d_in[25];
    const float* ins_v = (const float*)d_in[26]; const float* ins_g = (const float*)d_in[27]; const float* ins_b = (const float*)d_in[28];
    const float* prof_v = (const float*)d_in[29]; const float* prof_g = (const float*)d_in[30]; const float* prof_b = (const float*)d_in[31];
    const float* thr_v = (const float*)d_in[32]; const float* thr_g = (const float*)d_in[33]; const float* thr_b = (const float*)d_in[34];
    const float* idh_v = (const float*)d_in[35]; const float* idh_g = (const float*)d_in[36]; const float* idh_b = (const float*)d_in[37];
    const float* sev_v = (const float*)d_in[38]; const float* sev_g = (const float*)d_in[39]; const float* sev_b = (const float*)d_in[40];

    char* ws = (char*)d_ws;
    // Lifetime-overlapped workspace layout (~161 MB peak):
    float*    xg    = (float*)(ws);                         // [2][S][B][256] f32, written last
    uint16_t* a2    = (uint16_t*)(ws);                      // dead before xg written
    uint16_t* a1pad = (uint16_t*)(ws + 33554432);           // dead before a2's consumer done
    uint16_t* xpad  = (uint16_t*)(ws + 50364416);
    uint16_t* xfc   = (uint16_t*)(ws + 134217728);          // live while xg is written
    const size_t wb = 167772160;
    uint16_t* w1p   = (uint16_t*)(ws + wb);
    uint16_t* w2p   = (uint16_t*)(ws + wb + 49152);
    uint16_t* fcwp  = (uint16_t*)(ws + wb + 245760);
    uint16_t* wihfp = (uint16_t*)(ws + wb + 376832);
    uint16_t* wihbp = (uint16_t*)(ws + wb + 507904);
    uint16_t* whhfp = (uint16_t*)(ws + wb + 638976);
    uint16_t* whhbp = (uint16_t*)(ws + wb + 671744);
    float*    biasf = (float*)(ws + wb + 704512);
    float*    biasb = (float*)(ws + wb + 705536);
    float*    gmp   = (float*)(ws + wb + 706560);

    // ---- prep ----
    embed_kernel<<<16416, 256, 0, stream>>>(char_ids, emb, xpad);
    zero_a1_halo_kernel<<<64, 256, 0, stream>>>(a1pad);
    repack_conv1_kernel<<<96, 256, 0, stream>>>(conv1_w, w1p);
    repack_conv2_kernel<<<384, 256, 0, stream>>>(conv2_w, w2p);
    convert_bf16_kernel<<<256, 256, 0, stream>>>(fc_w, fcwp, 65536);
    convert_bf16_kernel<<<256, 256, 0, stream>>>(w_ih_f, wihfp, 65536);
    convert_bf16_kernel<<<256, 256, 0, stream>>>(w_ih_b, wihbp, 65536);
    convert_bf16_kernel<<<64, 256, 0, stream>>>(w_hh_f, whhfp, 16384);
    convert_bf16_kernel<<<64, 256, 0, stream>>>(w_hh_b, whhbp, 16384);
    bias_sum_kernel<<<2, 256, 0, stream>>>(b_ih_f, b_hh_f, b_ih_b, b_hh_b, biasf, biasb);

    // ---- conv1: M=128 K=192 (implicit im2col over padded x) ----
    gemm_wmma_kernel<true, true><<<dim3(1024, 8), 128, 0, stream>>>(
        w1p, xpad, conv1_b, a1pad + 128, nullptr,
        192, 64, 1026LL * 64, 1026LL * 128, 128LL);

    // ---- conv2: M=256 K=384 ----
    gemm_wmma_kernel<true, true><<<dim3(1024, 16), 128, 0, stream>>>(
        w2p, a1pad, conv2_b, a2, nullptr,
        384, 128, 1026LL * 128, 1024LL * 256, 256LL);

    // ---- fc: M=256 K=256 (no relu) ----
    gemm_wmma_kernel<true, false><<<dim3(1024, 16), 128, 0, stream>>>(
        fcwp, a2, fc_b, xfc, nullptr,
        256, 256, 1024LL * 256, 1024LL * 256, 256LL);

    // ---- LSTM input gates, both directions: out f32 in [s][b][g] order ----
    gemm_wmma_kernel<false, false><<<dim3(1024, 16), 128, 0, stream>>>(
        wihfp, xfc, biasf, nullptr, xg,
        256, 256, 1024LL * 256, 256LL, 64LL * 256);
    gemm_wmma_kernel<false, false><<<dim3(1024, 16), 128, 0, stream>>>(
        wihbp, xfc, biasb, nullptr, xg + (size_t)1024 * 64 * 256,
        256, 256, 1024LL * 256, 256LL, 64LL * 256);

    // ---- BiLSTM scan with fused global max-pool ----
    lstm_kernel<<<2, 256, 0, stream>>>(xg, whhfp, whhbp, gmp);

    // ---- heads ----
    head_kernel<<<1, 64, 0, stream>>>(gmp, tox_feat, feat_w, feat_b,
                                      bn_g, bn_b, bn_m, bn_v,
                                      tox_v, tox_g, tox_bb,
                                      ins_v, ins_g, ins_b,
                                      prof_v, prof_g, prof_b,
                                      thr_v, thr_g, thr_b,
                                      idh_v, idh_g, idh_b,
                                      sev_v, sev_g, sev_b,
                                      (float*)d_out);
}
